// GraphConv_68745246540466
// MI455X (gfx1250) — compile-verified, matrix operand order checked by reading, not run on hardware
//
#include <hip/hip_runtime.h>
#include <cstdint>
#include <cstddef>

typedef __bf16 bf16;
typedef __attribute__((ext_vector_type(16))) __bf16 v16bf;
typedef __attribute__((ext_vector_type(8)))  __bf16 v8bf;
typedef __attribute__((ext_vector_type(8)))  float   v8f;

__device__ __forceinline__ v16bf load_frag(const bf16* __restrict__ p)
{
  v8bf l0 = *(const v8bf*)(p);
  v8bf l1 = *(const v8bf*)(p + 16);
  v16bf f;
  #pragma unroll
  for (int i = 0; i < 8; ++i) { f[i] = l0[i]; f[8 + i] = l1[i]; }
  return f;
}

// ============================================================================
// GEMM:  C[M,N] (f32) = X[M,K] (bf16, row-major) * W[N,K]^T (bf16, row-major)
// wave32; each wave computes a 16 x (16*NT) strip (NT WMMA tiles), A fragment
// reused NT times.  Two-stage software pipeline (double-buffered fragments):
// stage k+1 loads are issued before stage k WMMAs, so per-wave WMMA issue
// overlaps memory latency instead of s_wait_loadcnt 0 on every tile.
// All tail guards are wave-uniform scalar branches (no acc phi-merges).
// K % 32 == 0, N % 16 == 0, M % 16 == 0 (guaranteed by this network).
// ============================================================================
template <int NT>
__global__ __launch_bounds__(256)
void gemm_nt_kernel(const bf16* __restrict__ X, const bf16* __restrict__ W,
                    float* __restrict__ C, int M, int N, int K, int nbase)
{
  const int wave = threadIdx.x >> 5;
  const int lane = threadIdx.x & 31;
  const int m0 = (blockIdx.x * 8 + wave) << 4;
  if (m0 >= M) return;                       // wave-uniform: EXEC stays full
  const int n0 = nbase + blockIdx.y * (16 * NT);
  const int hi = lane >> 4;                  // 0: K 0..7 & 16..23 ; 1: K 8..15 & 24..31
  const int lr = lane & 15;

  const bf16* __restrict__ xrow = X + (size_t)(m0 + lr) * K + hi * 8;
  const bf16* __restrict__ wrow = W + (size_t)(n0 + lr) * K + hi * 8;

  v8f acc[NT];
  #pragma unroll
  for (int t = 0; t < NT; ++t)
    acc[t] = v8f{0.f, 0.f, 0.f, 0.f, 0.f, 0.f, 0.f, 0.f};

  v16bf a0, a1;
  v16bf b0[NT], b1[NT];

  // preload stage 0
  a0 = load_frag(xrow);
  #pragma unroll
  for (int t = 0; t < NT; ++t) b0[t] = load_frag(wrow + (size_t)(t << 4) * K);

  for (int k0 = 0; k0 < K; k0 += 64) {
    const bool has1 = (k0 + 32 < K);
    if (has1) {                               // prefetch stage k0+32
      a1 = load_frag(xrow + k0 + 32);
      #pragma unroll
      for (int t = 0; t < NT; ++t)
        b1[t] = load_frag(wrow + (size_t)(t << 4) * K + k0 + 32);
    }
    #pragma unroll
    for (int t = 0; t < NT; ++t)              // consume stage k0
      acc[t] = __builtin_amdgcn_wmma_f32_16x16x32_bf16(
          false, a0, false, b0[t], (short)0, acc[t], false, false);
    if (k0 + 64 < K) {                        // prefetch stage k0+64
      a0 = load_frag(xrow + k0 + 64);
      #pragma unroll
      for (int t = 0; t < NT; ++t)
        b0[t] = load_frag(wrow + (size_t)(t << 4) * K + k0 + 64);
    }
    if (has1) {
      #pragma unroll
      for (int t = 0; t < NT; ++t)            // consume stage k0+32
        acc[t] = __builtin_amdgcn_wmma_f32_16x16x32_bf16(
            false, a1, false, b1[t], (short)0, acc[t], false, false);
    }
  }

  const int col = n0 + lr;
  const int rb  = m0 + hi * 8;
  #pragma unroll
  for (int t = 0; t < NT; ++t) {
    #pragma unroll
    for (int r = 0; r < 8; ++r)
      C[(size_t)(rb + r) * N + col + (t << 4)] = acc[t][r];
  }
}

// ============================================================================
// BatchNorm batch-statistics: per-column sum & sum-of-squares (atomic f32)
// ============================================================================
__global__ __launch_bounds__(256)
void colstats_kernel(const float* __restrict__ Y, float* __restrict__ s1,
                     float* __restrict__ s2, int M, int N)
{
  int r0 = blockIdx.x * 512;
  int r1 = r0 + 512; if (r1 > M) r1 = M;
  for (int c = threadIdx.x; c < N; c += 256) {
    float a = 0.f, b = 0.f;
    for (int r = r0; r < r1; ++r) {
      float v = Y[(size_t)r * N + c];
      a += v; b += v * v;
    }
    atomicAdd(&s1[c], a);
    atomicAdd(&s2[c], b);
  }
}

// BN (training-mode batch stats) + LeakyReLU; writes bf16 (strided slice) and/or f32
__global__ __launch_bounds__(256)
void bn_act_kernel(const float* __restrict__ Y,
                   const float* __restrict__ s1, const float* __restrict__ s2,
                   const float* __restrict__ g, const float* __restrict__ b,
                   bf16* __restrict__ outB, int bstride,
                   float* __restrict__ outF, int M, int N, float slope)
{
  size_t total = (size_t)M * N;
  float invM = 1.0f / (float)M;
  for (size_t i = (size_t)blockIdx.x * blockDim.x + threadIdx.x; i < total;
       i += (size_t)gridDim.x * blockDim.x) {
    int c = (int)(i % (size_t)N);
    float mu  = s1[c] * invM;
    float var = s2[c] * invM - mu * mu;
    float y = (Y[i] - mu) * rsqrtf(var + 1e-5f) * g[c] + b[c];
    y = (y > 0.f) ? y : slope * y;
    if (outB) { size_t r = i / (size_t)N; outB[r * (size_t)bstride + c] = (bf16)y; }
    if (outF) outF[i] = y;
  }
}

__global__ __launch_bounds__(256)
void fill_kernel(float* __restrict__ p, size_t n, float v)
{
  for (size_t i = (size_t)blockIdx.x * blockDim.x + threadIdx.x; i < n;
       i += (size_t)gridDim.x * blockDim.x) p[i] = v;
}

__global__ __launch_bounds__(256)
void cvt_kernel(const float* __restrict__ src, bf16* __restrict__ dst, size_t n)
{
  for (size_t i = (size_t)blockIdx.x * blockDim.x + threadIdx.x; i < n;
       i += (size_t)gridDim.x * blockDim.x) dst[i] = (bf16)src[i];
}

__global__ __launch_bounds__(256)
void copyb_kernel(const bf16* __restrict__ src, int sstride,
                  bf16* __restrict__ dst, int dstride, int cols, int rows)
{
  size_t total = (size_t)rows * cols;
  for (size_t i = (size_t)blockIdx.x * blockDim.x + threadIdx.x; i < total;
       i += (size_t)gridDim.x * blockDim.x) {
    size_t r = i / (size_t)cols; int c = (int)(i % (size_t)cols);
    dst[r * (size_t)dstride + c] = src[r * (size_t)sstride + c];
  }
}

// Edge-MLP input for StaticConv: [x_tgt, x_src - x_tgt, edge_attr] -> bf16
__global__ __launch_bounds__(256)
void gather_static_kernel(const float* __restrict__ x, int Dx,
                          const float* __restrict__ eattr, int De,
                          const int* __restrict__ src, const int* __restrict__ tgt,
                          bf16* __restrict__ out, int E)
{
  int Kin = 2 * Dx + De;
  size_t total = (size_t)E * Kin;
  for (size_t i = (size_t)blockIdx.x * blockDim.x + threadIdx.x; i < total;
       i += (size_t)gridDim.x * blockDim.x) {
    size_t e = i / (size_t)Kin; int c = (int)(i % (size_t)Kin);
    float v;
    if (c < Dx)            v = x[(size_t)tgt[e] * Dx + c];
    else if (c < 2 * Dx) { int cc = c - Dx;
                           v = x[(size_t)src[e] * Dx + cc] - x[(size_t)tgt[e] * Dx + cc]; }
    else                   v = eattr[e * (size_t)De + (c - 2 * Dx)];
    out[i] = (bf16)v;
  }
}

// Edge-MLP input for DynamicConv: [x_i, x_j - x_i] over kNN edges -> bf16
__global__ __launch_bounds__(256)
void gather_dyn_kernel(const float* __restrict__ x, int Dx,
                       const int* __restrict__ knn, int k,
                       bf16* __restrict__ out, int Nn)
{
  int Kin = 2 * Dx;
  size_t total = (size_t)Nn * k * Kin;
  for (size_t i = (size_t)blockIdx.x * blockDim.x + threadIdx.x; i < total;
       i += (size_t)gridDim.x * blockDim.x) {
    size_t r = i / (size_t)Kin; int c = (int)(i % (size_t)Kin);
    int ti = (int)(r / (size_t)k);
    int si = knn[r];
    float v = (c < Dx) ? x[(size_t)ti * Dx + c]
                       : x[(size_t)si * Dx + (c - Dx)] - x[(size_t)ti * Dx + (c - Dx)];
    out[i] = (bf16)v;
  }
}

__device__ __forceinline__ void atomicMaxF(float* addr, float val)
{
  unsigned int* ua = (unsigned int*)addr;
  unsigned int old = *ua;
  while (true) {
    float f = __uint_as_float(old);
    if (f >= val) break;
    unsigned int assumed = old;
    old = atomicCAS(ua, assumed, __float_as_uint(val));
    if (old == assumed) break;
  }
}

// scatter-max aggregation (tgt==nullptr => tgt = e / kdiv, for kNN edges)
__global__ __launch_bounds__(256)
void scatter_max_kernel(const float* __restrict__ msg, const int* __restrict__ tgt,
                        float* __restrict__ agg, int E, int N, int kdiv)
{
  size_t total = (size_t)E * N;
  for (size_t i = (size_t)blockIdx.x * blockDim.x + threadIdx.x; i < total;
       i += (size_t)gridDim.x * blockDim.x) {
    size_t e = i / (size_t)N; int c = (int)(i % (size_t)N);
    int t = tgt ? tgt[e] : (int)(e / (size_t)kdiv);
    atomicMaxF(&agg[(size_t)t * N + c], msg[i]);
  }
}

// out = lin + (finite(agg) ? agg : 0); writes bf16 slice (into cat1) and opt. f32
__global__ __launch_bounds__(256)
void add_agg_kernel(const float* __restrict__ lin, const float* __restrict__ agg,
                    bf16* __restrict__ outB, int bstride,
                    float* __restrict__ outF, int M, int N)
{
  size_t total = (size_t)M * N;
  for (size_t i = (size_t)blockIdx.x * blockDim.x + threadIdx.x; i < total;
       i += (size_t)gridDim.x * blockDim.x) {
    float a = agg[i];
    if (!(a > -1e37f)) a = 0.f;             // empty segment -> 0
    float v = lin[i] + a;
    size_t r = i / (size_t)N; int c = (int)(i % (size_t)N);
    outB[r * (size_t)bstride + c] = (bf16)v;
    if (outF) outF[i] = v;
  }
}

// ============================================================================
// kNN (squared euclidean, self-loops kept): one wave32 per target node.
// Per-lane top-16 in registers, then 16 rounds of wave argmin via __shfl_xor.
// Feature matrix is 2.5-5 MB -> L2-resident on MI455X (192 MB L2).
// ============================================================================
#define KNN_K 16
__global__ __launch_bounds__(256)
void knn_kernel(const float* __restrict__ x, int Dx, int Nn, int* __restrict__ knn)
{
  __shared__ float s_xi[8 * 128];
  const int wave = threadIdx.x >> 5;
  const int lane = threadIdx.x & 31;
  const int i = blockIdx.x * 8 + wave;
  const bool valid = (i < Nn);
  float* xi = s_xi + wave * 128;
  if (valid) for (int c = lane; c < Dx; c += 32) xi[c] = x[(size_t)i * Dx + c];
  __syncthreads();
  if (!valid) return;

  float dl[KNN_K]; int il[KNN_K];
  #pragma unroll
  for (int s = 0; s < KNN_K; ++s) { dl[s] = 3.4e38f; il[s] = 0x7fffffff; }
  float worst = 3.4e38f; int wslot = 0;

  for (int j = lane; j < Nn; j += 32) {
    const float* xj = x + (size_t)j * Dx;
    float d = 0.f;
    for (int c = 0; c < Dx; ++c) { float t = xi[c] - xj[c]; d += t * t; }
    if (d < worst) {
      dl[wslot] = d; il[wslot] = j;
      worst = -1.f;
      #pragma unroll
      for (int s = 0; s < KNN_K; ++s)
        if (dl[s] > worst) { worst = dl[s]; wslot = s; }
    }
  }

  for (int r = 0; r < KNN_K; ++r) {
    float bd = 3.4e38f; int bs = -1;
    #pragma unroll
    for (int s = 0; s < KNN_K; ++s)
      if (dl[s] < bd) { bd = dl[s]; bs = s; }
    int bj = (bs >= 0) ? il[bs] : 0x7fffffff;
    for (int off = 16; off > 0; off >>= 1) {
      float od = __shfl_xor(bd, off);
      int   oj = __shfl_xor(bj, off);
      if (od < bd || (od == bd && oj < bj)) { bd = od; bj = oj; }
    }
    if (bs >= 0 && il[bs] == bj) { dl[bs] = 3.4e38f; il[bs] = 0x7fffffff; }
    if (lane == 0) knn[(size_t)i * KNN_K + r] = bj;
  }
}

// ============================================================================
// Host orchestration
// ============================================================================
struct Lay { const bf16* Wb; const float* g; const float* b; int n; int k; };

extern "C" void kernel_launch(void* const* d_in, const int* in_sizes, int n_in,
                              void* d_out, int out_size, void* d_ws, size_t ws_size,
                              hipStream_t stream)
{
  (void)n_in; (void)out_size; (void)ws_size;
  const int D  = 64;
  const int Nn = in_sizes[0] / D;         // 10000
  const int E  = in_sizes[2] / 2;         // 320000
  const int K  = 16;
  const int Ed = Nn * K;                  // 160000 dynamic edges

  const float* node  = (const float*)d_in[0];
  const float* eattr = (const float*)d_in[1];
  const int*   src   = (const int*)d_in[2];
  const int*   tgt   = src + E;

  // ---- bump allocator over workspace ----
  char* base = (char*)d_ws;
  size_t off = 0;
  auto alloc = [&](size_t bytes) -> void* {
    void* r = base + off;
    off = (off + bytes + 255) & ~(size_t)255;
    return r;
  };

  float* stats  = (float*)alloc(4096 * sizeof(float));
  float* stat_s = stats, *stat_ss = stats + 2048;
  float* x_sg1 = (float*)alloc((size_t)Nn * 64  * 4);
  float* x_sg2 = (float*)alloc((size_t)Nn * 128 * 4);
  float* x_dg1 = (float*)alloc((size_t)Nn * 128 * 4);
  bf16*  cat1  = (bf16*) alloc((size_t)Nn * 704 * 2);
  bf16*  cat2  = (bf16*) alloc((size_t)Nn * 1728 * 2);
  int*   knn   = (int*)  alloc((size_t)Nn * K * 4);
  bf16*  Ein   = (bf16*) alloc((size_t)E * 320 * 2);
  float* Ea    = (float*)alloc((size_t)E * 256 * 4);
  bf16*  Eb    = (bf16*) alloc((size_t)E * 256 * 2);
  float* agg   = (float*)alloc((size_t)Nn * 192 * 4);
  float* linY  = (float*)alloc((size_t)Nn * 1024 * 4);
  bf16*  linB  = (bf16*) alloc((size_t)Nn * 1024 * 2);
  bf16*  xb    = (bf16*) alloc((size_t)Nn * 128 * 2);

  auto gs = [](size_t total) -> unsigned {
    size_t b = (total + 255) / 256;
    if (b < 1) b = 1;
    if (b > 32768) b = 32768;
    return (unsigned)b;
  };

  // GEMM dispatch: main NT=4 launch + optional NT=1..3 tail (only N=160 needs it)
  auto runGemm = [&](const bf16* X, const bf16* W, float* C, int M, int N, int Kd) {
    int nfull = N >> 6;                   // chunks of 64 columns
    dim3 gm((M + 127) / 128, nfull);
    if (nfull > 0)
      gemm_nt_kernel<4><<<gm, 256, 0, stream>>>(X, W, C, M, N, Kd, 0);
    int rem = N - (nfull << 6);
    if (rem > 0) {
      dim3 gt((M + 127) / 128, 1);
      switch (rem >> 4) {
        case 1: gemm_nt_kernel<1><<<gt, 256, 0, stream>>>(X, W, C, M, N, Kd, nfull << 6); break;
        case 2: gemm_nt_kernel<2><<<gt, 256, 0, stream>>>(X, W, C, M, N, Kd, nfull << 6); break;
        case 3: gemm_nt_kernel<3><<<gt, 256, 0, stream>>>(X, W, C, M, N, Kd, nfull << 6); break;
      }
    }
  };

  // convert a weight matrix (f32 [n,k]) to bf16 in workspace
  auto cvtW = [&](int wi, int n, int k) -> const bf16* {
    bf16* wb = (bf16*)alloc((size_t)n * k * sizeof(bf16));
    cvt_kernel<<<gs((size_t)n * k), 256, 0, stream>>>((const float*)d_in[wi], wb, (size_t)n * k);
    return wb;
  };
  auto mkLay = [&](int wi, int n, int k) -> Lay {
    return Lay{ cvtW(wi, n, k), (const float*)d_in[wi + 1], (const float*)d_in[wi + 2], n, k };
  };

  // jax pytree flattening of params (dicts: alphabetical keys)
  Lay dg1_lin[2] = { mkLay(4, 128, 64),   mkLay(7, 128, 128) };
  Lay dg1_msg[2] = { mkLay(10, 128, 128), mkLay(13, 128, 128) };
  Lay dg2_lin[2] = { mkLay(16, 192, 128), mkLay(19, 192, 192) };
  Lay dg2_msg[2] = { mkLay(22, 192, 256), mkLay(25, 192, 192) };
  Lay fuse1      =   mkLay(28, 1024, 704);
  const bf16* WoutB = cvtW(31, 128, 256);
  Lay f2a        =   mkLay(32, 512, 1728);
  Lay f2b        =   mkLay(35, 256, 512);
  Lay sg1_lin[2] = { mkLay(38, 64, 64),   mkLay(41, 64, 64) };
  Lay sg1_msg[3] = { mkLay(44, 128, 192), mkLay(47, 64, 128),  mkLay(50, 64, 64) };
  Lay sg2_lin[2] = { mkLay(53, 128, 64),  mkLay(56, 128, 128) };
  Lay sg2_msg[3] = { mkLay(59, 160, 192), mkLay(62, 128, 160), mkLay(65, 128, 128) };
  Lay sg3_lin[2] = { mkLay(68, 192, 128), mkLay(71, 192, 192) };
  Lay sg3_msg[3] = { mkLay(74, 256, 320), mkLay(77, 192, 256), mkLay(80, 192, 192) };

  // Linear -> BN(batch stats) -> LeakyReLU
  auto runLayer = [&](const bf16* Xb2, int M, const Lay& L, float* Y,
                      bf16* outB, int bstride, float* outF, float slope) {
    runGemm(Xb2, L.Wb, Y, M, L.n, L.k);
    fill_kernel<<<gs(4096), 256, 0, stream>>>(stats, 4096, 0.f);
    colstats_kernel<<<(M + 511) / 512, 256, 0, stream>>>(Y, stat_s, stat_ss, M, L.n);
    bn_act_kernel<<<gs((size_t)M * L.n), 256, 0, stream>>>(
        Y, stat_s, stat_ss, L.g, L.b, outB, bstride, outF, M, L.n, slope);
  };

  auto staticConv = [&](const float* xin, int Dx, const Lay* lin, const Lay* msg, int nmsg,
                        bf16* catSlice, int cstride, float* xoutF, float slope) {
    int Nout = msg[nmsg - 1].n;
    int Kin  = 2 * Dx + D;
    gather_static_kernel<<<gs((size_t)E * Kin), 256, 0, stream>>>(
        xin, Dx, eattr, D, src, tgt, Ein, E);
    const bf16* Xc = Ein;
    for (int l = 0; l < nmsg; ++l) {
      bool last = (l == nmsg - 1);
      runLayer(Xc, E, msg[l], Ea, last ? nullptr : Eb, msg[l].n, last ? Ea : nullptr, slope);
      Xc = Eb;
    }
    fill_kernel<<<gs((size_t)Nn * Nout), 256, 0, stream>>>(agg, (size_t)Nn * Nout, -3.3e38f);
    scatter_max_kernel<<<gs((size_t)E * Nout), 256, 0, stream>>>(Ea, tgt, agg, E, Nout, 1);
    cvt_kernel<<<gs((size_t)Nn * Dx), 256, 0, stream>>>(xin, xb, (size_t)Nn * Dx);
    runLayer(xb, Nn, lin[0], linY, linB, lin[0].n, nullptr, slope);
    runLayer(linB, Nn, lin[1], linY, nullptr, 0, linY, slope);
    add_agg_kernel<<<gs((size_t)Nn * Nout), 256, 0, stream>>>(
        linY, agg, catSlice, cstride, xoutF, Nn, Nout);
  };

  auto dynConv = [&](const float* xin, int Dx, const Lay* lin, const Lay* msg,
                     bf16* catSlice, int cstride, float* xoutF, float slope) {
    int Nout = msg[1].n;
    int Kin  = 2 * Dx;
    knn_kernel<<<(Nn + 7) / 8, 256, 0, stream>>>(xin, Dx, Nn, knn);
    gather_dyn_kernel<<<gs((size_t)Ed * Kin), 256, 0, stream>>>(xin, Dx, knn, K, Ein, Nn);
    const bf16* Xc = Ein;
    for (int l = 0; l < 2; ++l) {
      bool last = (l == 1);
      runLayer(Xc, Ed, msg[l], Ea, last ? nullptr : Eb, msg[l].n, last ? Ea : nullptr, slope);
      Xc = Eb;
    }
    fill_kernel<<<gs((size_t)Nn * Nout), 256, 0, stream>>>(agg, (size_t)Nn * Nout, -3.3e38f);
    scatter_max_kernel<<<gs((size_t)Ed * Nout), 256, 0, stream>>>(Ea, nullptr, agg, Ed, Nout, K);
    cvt_kernel<<<gs((size_t)Nn * Dx), 256, 0, stream>>>(xin, xb, (size_t)Nn * Dx);
    runLayer(xb, Nn, lin[0], linY, linB, lin[0].n, nullptr, slope);
    runLayer(linB, Nn, lin[1], linY, nullptr, 0, linY, slope);
    add_agg_kernel<<<gs((size_t)Nn * Nout), 256, 0, stream>>>(
        linY, agg, catSlice, cstride, xoutF, Nn, Nout);
  };

  // cat1 layout: [sg1 (0..64) | dg1 (64..192) | dg2 (192..384) | sg2 (384..512) | sg3 (512..704)]
  staticConv(node,  64,  sg1_lin, sg1_msg, 3, cat1 + 0,   704, x_sg1,   0.1f);
  staticConv(x_sg1, 64,  sg2_lin, sg2_msg, 3, cat1 + 384, 704, x_sg2,   0.1f);
  staticConv(x_sg2, 128, sg3_lin, sg3_msg, 3, cat1 + 512, 704, nullptr, 0.1f);
  dynConv  (x_sg1, 64,  dg1_lin, dg1_msg,    cat1 + 64,  704, x_dg1,   0.2f);
  dynConv  (x_dg1, 128, dg2_lin, dg2_msg,    cat1 + 192, 704, nullptr, 0.2f);

  // fuse1: f1 = MLP(cat1) -> cat2[:, 0:1024]
  runLayer(cat1, Nn, fuse1, linY, cat2, 1728, nullptr, 0.2f);
  // cat2[:, 1024:1728] = cat1
  copyb_kernel<<<gs((size_t)Nn * 704), 256, 0, stream>>>(cat1, 704, cat2 + 1024, 1728, 704, Nn);
  // fuse2: 1728 -> 512 -> 256
  runLayer(cat2, Nn, f2a, linY, linB, 512, nullptr, 0.2f);
  runLayer(linB, Nn, f2b, linY, linB, 256, nullptr, 0.2f);
  // final projection (no BN): out = h @ Wout^T  -> f32 d_out [Nn, 128]
  runGemm(linB, WoutB, (float*)d_out, Nn, 128, 256);
}
